// ContextQueryAttention_71416716198066
// MI455X (gfx1250) — matrix operand order repaired and checked by reading, not compiled, not run on hardware
//
#include <hip/hip_runtime.h>

typedef _Float16 half_t;
typedef __attribute__((ext_vector_type(16))) _Float16 v16h;
typedef __attribute__((ext_vector_type(8)))  _Float16 v8h;
typedef __attribute__((ext_vector_type(8)))  float    v8f;

#define BATCH 64
#define NCTX  1024
#define MQ    128
#define DDIM  128
#define NTILE 128
#define NT    (NCTX / NTILE)   // 8 row tiles per batch

static __device__ __forceinline__ v8f wmma_f16(v16h a, v16h b, v8f c) {
  // D(f32 16x16) = A(f16 16x32) x B(f16 32x16) + C
  return __builtin_amdgcn_wmma_f32_16x16x32_f16(false, a, false, b, (short)0, c, false, false);
}

// ---------------------------------------------------------------------------
// Kernel 1: per (batch, 128-row tile):
//   S = C @ Qhat^T + qw2,  Qhat = w3*Q + w1,  qw2 = Q @ w2
//   Pu = exp(S - rowmax)  (f16, stored row-major and transposed)
//   row stats (rowmax, rowsum), tile column stats (max, sumexp)
// ---------------------------------------------------------------------------
__global__ __launch_bounds__(256)
void kernelS(const float* __restrict__ Cg, const float* __restrict__ Qg,
             const float* __restrict__ Wg,
             half_t* __restrict__ P, half_t* __restrict__ Pt,
             float* __restrict__ rmaxg, float* __restrict__ rsumg,
             float* __restrict__ colst)
{
  const int b   = blockIdx.x >> 3;
  const int t   = blockIdx.x & 7;
  const int n0  = t * NTILE;
  const int tid = threadIdx.x;
  const int lane = tid & 31;
  const int wave = tid >> 5;
  const int l16  = lane & 15;
  const bool lo  = lane < 16;

  __shared__ half_t qhat[128][136];   // Qhat (f16, padded); reused to stage Pu
  __shared__ float  qw2s[128];
  __shared__ float  red[8][128];
  __shared__ float  gcm[128];

  const float* Cb = Cg + (size_t)b * NCTX * DDIM;
  const float* Qb = Qg + (size_t)b * MQ * DDIM;

  // Stage qhat[m][d] = Q[m][d]*w3[d] + w1[d]
  for (int idx = tid; idx < 128 * 32; idx += 256) {
    int m  = idx >> 5;
    int d4 = (idx & 31) << 2;
    float4 q  = *(const float4*)(Qb + m * DDIM + d4);
    float4 w1 = *(const float4*)(Wg + d4);
    float4 w3 = *(const float4*)(Wg + 2 * DDIM + d4);
    qhat[m][d4 + 0] = (half_t)(q.x * w3.x + w1.x);
    qhat[m][d4 + 1] = (half_t)(q.y * w3.y + w1.y);
    qhat[m][d4 + 2] = (half_t)(q.z * w3.z + w1.z);
    qhat[m][d4 + 3] = (half_t)(q.w * w3.w + w1.w);
  }
  if (tid < 128) {
    float s = 0.f;
    for (int d = 0; d < DDIM; ++d) s += Qb[tid * DDIM + d] * Wg[DDIM + d];
    qw2s[tid] = s;
  }
  __syncthreads();

  v8f acc[8] = {};
  const int arow = (wave << 4) + l16;        // A-matrix row for this lane
  const int kbA  = lo ? 0 : 8;               // A-frag K sub-block base
  const int kbB  = lo ? 0 : 8;               // B-frag K base
  const float* crow = Cb + (size_t)(n0 + arow) * DDIM;

  for (int kc = 0; kc < 4; ++kc) {           // K = 128 in chunks of 32
    const float* pa = crow + kc * 32 + kbA;
    v16h a;
    #pragma unroll
    for (int h = 0; h < 8; ++h) { a[h] = (half_t)pa[h]; a[8 + h] = (half_t)pa[16 + h]; }
    #pragma unroll
    for (int j = 0; j < 8; ++j) {
      const half_t* pb = &qhat[j * 16 + l16][kc * 32 + kbB];
      v8h b0 = *(const v8h*)pb;
      v8h b1 = *(const v8h*)(pb + 16);
      v16h bf;
      #pragma unroll
      for (int h = 0; h < 8; ++h) { bf[h] = b0[h]; bf[8 + h] = b1[h]; }
      acc[j] = wmma_f16(a, bf, acc[j]);
    }
  }

  // add qw2[col] bias
  #pragma unroll
  for (int j = 0; j < 8; ++j) {
    float bias = qw2s[j * 16 + l16];
    #pragma unroll
    for (int i = 0; i < 8; ++i) acc[j][i] += bias;
  }

  // ---- tile column stats (over this tile's 128 rows) ----
  #pragma unroll
  for (int j = 0; j < 8; ++j) {
    float m = acc[j][0];
    #pragma unroll
    for (int i = 1; i < 8; ++i) m = fmaxf(m, acc[j][i]);
    m = fmaxf(m, __shfl_xor(m, 16));        // merge rows 0-7 with 8-15
    if (lo) red[wave][j * 16 + l16] = m;
  }
  __syncthreads();
  if (tid < 128) {
    float m = red[0][tid];
    for (int w = 1; w < 8; ++w) m = fmaxf(m, red[w][tid]);
    gcm[tid] = m;
  }
  __syncthreads();
  #pragma unroll
  for (int j = 0; j < 8; ++j) {
    float g = gcm[j * 16 + l16];
    float s = 0.f;
    #pragma unroll
    for (int i = 0; i < 8; ++i) s += __expf(acc[j][i] - g);
    s += __shfl_xor(s, 16);
    if (lo) red[wave][j * 16 + l16] = s;
  }

  // ---- row stats (full row lives in one 16-lane half: masks 1..8) ----
  float rmx[8], rsm[8];
  #pragma unroll
  for (int i = 0; i < 8; ++i) {
    float m = acc[0][i];
    #pragma unroll
    for (int j = 1; j < 8; ++j) m = fmaxf(m, acc[j][i]);
    #pragma unroll
    for (int msk = 8; msk >= 1; msk >>= 1) m = fmaxf(m, __shfl_xor(m, msk));
    float s = 0.f;
    #pragma unroll
    for (int j = 0; j < 8; ++j) s += __expf(acc[j][i] - m);
    #pragma unroll
    for (int msk = 8; msk >= 1; msk >>= 1) s += __shfl_xor(s, msk);
    rmx[i] = m; rsm[i] = s;
  }
  const int rlo = (wave << 4) + (lo ? 0 : 8);   // D-matrix row base for this lane
  if (l16 == 0) {
    #pragma unroll
    for (int i = 0; i < 8; ++i) {
      rmaxg[(size_t)b * NCTX + n0 + rlo + i] = rmx[i];
      rsumg[(size_t)b * NCTX + n0 + rlo + i] = rsm[i];
    }
  }
  // Pu = exp(S - rowmax)  in-place
  #pragma unroll
  for (int j = 0; j < 8; ++j)
    #pragma unroll
    for (int i = 0; i < 8; ++i)
      acc[j][i] = __expf(acc[j][i] - rmx[i]);

  __syncthreads();
  if (tid < 128) {
    float s = 0.f;
    for (int w = 0; w < 8; ++w) s += red[w][tid];
    float* cs = colst + (((size_t)b * 8 + t) * 128 + tid) * 2;
    cs[0] = gcm[tid];
    cs[1] = s;
  }
  // stage Pu into LDS (reuse qhat), then vectorized global writes
  #pragma unroll
  for (int j = 0; j < 8; ++j)
    #pragma unroll
    for (int i = 0; i < 8; ++i)
      qhat[rlo + i][j * 16 + l16] = (half_t)acc[j][i];
  __syncthreads();

  for (int idx = tid; idx < 128 * 16; idx += 256) {
    int r  = idx >> 4;
    int c8 = (idx & 15) << 3;
    *(uint4*)(P + ((size_t)b * NCTX + n0 + r) * MQ + c8) = *(const uint4*)&qhat[r][c8];
  }
  for (int idx = tid; idx < 128 * 16; idx += 256) {
    int m  = idx >> 4;
    int c8 = (idx & 15) << 3;
    union { half_t h[8]; uint4 u; } tmp;
    #pragma unroll
    for (int k = 0; k < 8; ++k) tmp.h[k] = qhat[c8 + k][m];
    *(uint4*)(Pt + ((size_t)b * MQ + m) * NCTX + n0 + c8) = tmp.u;
  }
}

// ---------------------------------------------------------------------------
// Kernel 2: per batch:  T[m][d] = sum_n S_col[n][m] * C[n][d], stored as T^T
//   S_col = Pu * exp(rowmax[n] - M[m]) / L[m]  (elementwise <= 1 before f16)
// ---------------------------------------------------------------------------
__global__ __launch_bounds__(256)
void kernelT(const float* __restrict__ Cg, const half_t* __restrict__ Pt,
             const float* __restrict__ rmaxg, const float* __restrict__ colst,
             half_t* __restrict__ Tt)
{
  const int b = blockIdx.x;
  const int tid = threadIdx.x, lane = tid & 31, wave = tid >> 5;
  const int l16 = lane & 15;
  const bool lo = lane < 16;

  __shared__ float Mm[128], Ll[128], emM[128];
  __shared__ float ern[1024];
  __shared__ float gshared;
  __shared__ half_t ct[128][136];   // C^T tile (f16); reused to stage T

  const float* Cb = Cg + (size_t)b * NCTX * DDIM;

  if (tid < 128) {
    float M = -3.4e38f;
    float mt[8], lt[8];
    #pragma unroll
    for (int t = 0; t < 8; ++t) {
      const float* cs = colst + (((size_t)b * 8 + t) * 128 + tid) * 2;
      mt[t] = cs[0]; lt[t] = cs[1];
      M = fmaxf(M, mt[t]);
    }
    float L = 0.f;
    #pragma unroll
    for (int t = 0; t < 8; ++t) L += lt[t] * __expf(mt[t] - M);
    Mm[tid] = M; Ll[tid] = L;
  }
  __syncthreads();
  if (tid == 0) {
    float g = Mm[0];
    for (int i = 1; i < 128; ++i) g = fmaxf(g, Mm[i]);
    gshared = g;
  }
  __syncthreads();
  const float G = gshared;
  if (tid < 128) emM[tid] = __expf(G - Mm[tid]) / Ll[tid];
  for (int idx = tid; idx < 1024; idx += 256)
    ern[idx] = __expf(rmaxg[(size_t)b * NCTX + idx] - G);
  __syncthreads();

  v8f acc[8] = {};
  const int mrow = (wave << 4) + l16;
  const float escale = emM[mrow];
  const int kbA = lo ? 0 : 8;
  const int kbB = lo ? 0 : 8;
  const half_t* ptrow = Pt + ((size_t)b * MQ + mrow) * NCTX;

  for (int t = 0; t < 8; ++t) {
    const int n0 = t * NTILE;
    __syncthreads();
    for (int idx = tid; idx < 128 * 32; idx += 256) {   // stage C^T f16
      int r  = idx >> 5;
      int c4 = (idx & 31) << 2;
      float4 v = *(const float4*)(Cb + (size_t)(n0 + r) * DDIM + c4);
      ct[c4 + 0][r] = (half_t)v.x;
      ct[c4 + 1][r] = (half_t)v.y;
      ct[c4 + 2][r] = (half_t)v.z;
      ct[c4 + 3][r] = (half_t)v.w;
    }
    __syncthreads();
    #pragma unroll
    for (int kc = 0; kc < 4; ++kc) {
      const int base = n0 + kc * 32 + kbA;
      const half_t* pa = ptrow + base;
      v16h a;
      #pragma unroll
      for (int h = 0; h < 8; ++h) {
        a[h]     = (half_t)((float)pa[h]      * ern[base + h]      * escale);
        a[8 + h] = (half_t)((float)pa[16 + h] * ern[base + 16 + h] * escale);
      }
      #pragma unroll
      for (int j = 0; j < 8; ++j) {
        const half_t* pb = &ct[j * 16 + l16][kc * 32 + kbB];
        v8h b0 = *(const v8h*)pb;
        v8h b1 = *(const v8h*)(pb + 16);
        v16h bf;
        #pragma unroll
        for (int h = 0; h < 8; ++h) { bf[h] = b0[h]; bf[8 + h] = b1[h]; }
        acc[j] = wmma_f16(a, bf, acc[j]);
      }
    }
  }
  __syncthreads();
  const int rlo = (wave << 4) + (lo ? 0 : 8);
  #pragma unroll
  for (int j = 0; j < 8; ++j)
    #pragma unroll
    for (int i = 0; i < 8; ++i)
      ct[rlo + i][j * 16 + l16] = (half_t)acc[j][i];   // T[m][d]
  __syncthreads();
  for (int idx = tid; idx < 128 * 16; idx += 256) {    // write T^T[d][m]
    int d  = idx >> 4;
    int c8 = (idx & 15) << 3;
    union { half_t h[8]; uint4 u; } tmp;
    #pragma unroll
    for (int k = 0; k < 8; ++k) tmp.h[k] = ct[c8 + k][d];
    *(uint4*)(Tt + ((size_t)b * MQ + d) * MQ + c8) = tmp.u;
  }
}

// ---------------------------------------------------------------------------
// Kernel 3: per (batch, tile):  A = (Pu/rowsum)@Q, B = (Pu/rowsum)@T,
//   out = [C, A, C*A, C*B]
// ---------------------------------------------------------------------------
__global__ __launch_bounds__(256)
void kernelOut(const float* __restrict__ Cg, const float* __restrict__ Qg,
               const half_t* __restrict__ P, const half_t* __restrict__ Tt,
               const float* __restrict__ rsumg, float* __restrict__ Og)
{
  const int b  = blockIdx.x >> 3;
  const int t  = blockIdx.x & 7;
  const int n0 = t * NTILE;
  const int tid = threadIdx.x, lane = tid & 31, wave = tid >> 5;
  const int l16 = lane & 15;
  const bool lo = lane < 16;

  __shared__ half_t qT[128][136];
  __shared__ float  rs[128];

  const float* Cb = Cg + (size_t)b * NCTX * DDIM;
  const float* Qb = Qg + (size_t)b * MQ * DDIM;
  for (int idx = tid; idx < 128 * 32; idx += 256) {    // stage Q^T f16
    int m  = idx >> 5;
    int d4 = (idx & 31) << 2;
    float4 v = *(const float4*)(Qb + m * DDIM + d4);
    qT[d4 + 0][m] = (half_t)v.x;
    qT[d4 + 1][m] = (half_t)v.y;
    qT[d4 + 2][m] = (half_t)v.z;
    qT[d4 + 3][m] = (half_t)v.w;
  }
  if (tid < 128) rs[tid] = rsumg[(size_t)b * NCTX + n0 + tid];
  __syncthreads();

  v8f accA[8] = {}, accB[8] = {};
  const int arow = (wave << 4) + l16;
  const int kbA = lo ? 0 : 8;
  const int kbB = lo ? 0 : 8;
  const half_t* prow = P + ((size_t)b * NCTX + n0 + arow) * MQ;
  const half_t* ttb  = Tt + (size_t)b * MQ * MQ;

  #pragma unroll
  for (int kc = 0; kc < 4; ++kc) {
    const half_t* pa = prow + kc * 32 + kbA;
    v8h a0 = *(const v8h*)pa;
    v8h a1 = *(const v8h*)(pa + 16);
    v16h a;
    #pragma unroll
    for (int h = 0; h < 8; ++h) { a[h] = a0[h]; a[8 + h] = a1[h]; }
    #pragma unroll
    for (int j = 0; j < 8; ++j) {
      const int col = j * 16 + l16;
      const half_t* pq = &qT[col][kc * 32 + kbB];
      v8h q0 = *(const v8h*)pq; v8h q1 = *(const v8h*)(pq + 16);
      v16h bq;
      #pragma unroll
      for (int h = 0; h < 8; ++h) { bq[h] = q0[h]; bq[8 + h] = q1[h]; }
      accA[j] = wmma_f16(a, bq, accA[j]);
      const half_t* pt = ttb + (size_t)col * MQ + kc * 32 + kbB;
      v8h t0 = *(const v8h*)pt; v8h t1 = *(const v8h*)(pt + 16);
      v16h bt;
      #pragma unroll
      for (int h = 0; h < 8; ++h) { bt[h] = t0[h]; bt[8 + h] = t1[h]; }
      accB[j] = wmma_f16(a, bt, accB[j]);
    }
  }

  const int rlo = (wave << 4) + (lo ? 0 : 8);
  float inv[8];
  #pragma unroll
  for (int i = 0; i < 8; ++i) inv[i] = 1.0f / rs[rlo + i];

  #pragma unroll
  for (int j = 0; j < 8; ++j) {
    const int col = j * 16 + l16;
    #pragma unroll
    for (int i = 0; i < 8; ++i) {
      const int n = n0 + rlo + i;
      float c  = Cb[(size_t)n * DDIM + col];
      float av = accA[j][i] * inv[i];
      float bv = accB[j][i] * inv[i];
      float* o = Og + ((size_t)b * NCTX + n) * (4 * DDIM);
      o[col]            = c;
      o[DDIM + col]     = av;
      o[2 * DDIM + col] = c * av;
      o[3 * DDIM + col] = c * bv;
    }
  }
}

// ---------------------------------------------------------------------------
extern "C" void kernel_launch(void* const* d_in, const int* in_sizes, int n_in,
                              void* d_out, int out_size, void* d_ws, size_t ws_size,
                              hipStream_t stream)
{
  (void)in_sizes; (void)n_in; (void)out_size; (void)ws_size;
  const float* C = (const float*)d_in[0];
  const float* Q = (const float*)d_in[1];
  const float* W = (const float*)d_in[2];
  float* out = (float*)d_out;

  char* ws = (char*)d_ws;
  half_t* P     = (half_t*)(ws);                               // 16 MB
  half_t* Pt    = (half_t*)(ws + ((size_t)16 << 20));          // 16 MB
  half_t* Tt    = (half_t*)(ws + ((size_t)32 << 20));          //  2 MB
  float*  rmaxg = (float*) (ws + ((size_t)34 << 20));          // 256 KB
  float*  rsumg = (float*) (ws + ((size_t)34 << 20) + ((size_t)256 << 10));
  float*  colst = (float*) (ws + ((size_t)34 << 20) + ((size_t)512 << 10)); // 512 KB

  kernelS  <<<dim3(BATCH * NT), dim3(256), 0, stream>>>(C, Q, W, P, Pt, rmaxg, rsumg, colst);
  kernelT  <<<dim3(BATCH),      dim3(256), 0, stream>>>(C, Pt, rmaxg, colst, Tt);
  kernelOut<<<dim3(BATCH * NT), dim3(256), 0, stream>>>(C, Q, P, Tt, rsumg, out);
}
